// Attention_45638322487970
// MI455X (gfx1250) — matrix operand order repaired and checked by reading, not compiled, not run on hardware
//
#include <hip/hip_runtime.h>

// ---------------------------------------------------------------------------
// Attention layer for MI455X (gfx1250): bf16 WMMA everywhere, f32 accumulate.
// B=2, S=2048, HID=2048, NH=16, NKV=8, HD=128.
// Adds gfx1250 async global->LDS copies (ASYNCcnt) for pure-copy staging.
// ---------------------------------------------------------------------------

#define B_   2
#define S_   2048
#define HID_ 2048
#define NH_  16
#define NKV_ 8
#define HD_  128

typedef __attribute__((ext_vector_type(16))) __bf16 bf16x16;
typedef __attribute__((ext_vector_type(8)))  float  f32x8;

union FragB { bf16x16 v; uint4 q[2]; __bf16 e[16]; };
union AccF  { f32x8  v; float e[8]; };

// -------- gfx1250 async global->LDS copy (ASYNCcnt-tracked) ----------------
// GLOBAL_LOAD_ASYNC_TO_LDS_B128: VDST = per-lane LDS byte address, VADDR =
// 64-bit global address (GV mode, SADDR=off). Generic pointers to LDS carry
// the LDS offset in their low 32 bits.
__device__ __forceinline__ void async_g2lds_b128(void* lds, const void* g) {
  unsigned int l = (unsigned int)(unsigned long long)lds;
  asm volatile("global_load_async_to_lds_b128 %0, %1, off"
               :: "v"(l), "v"(g) : "memory");
}
__device__ __forceinline__ void wait_async0() {
  asm volatile("s_wait_asynccnt 0" ::: "memory");
}

// A-matrix fragment, 16x32 bf16, row-major source with given element stride.
// ISA layout: lanes 0-15 hold rows 0-15 with K = {0..7, 16..23};
//             lanes 16-31 hold rows 0-15 with K = {8..15, 24..31}.
__device__ __forceinline__ bf16x16 load_frag_a(const __bf16* base, int stride) {
  const int lane = threadIdx.x & 31;
  const int r    = lane & 15;
  const int klo  = (lane < 16) ? 0 : 8;
  const __bf16* p = base + r * stride + klo;
  FragB u;
  u.q[0] = *reinterpret_cast<const uint4*>(p);        // K = klo .. klo+7
  u.q[1] = *reinterpret_cast<const uint4*>(p + 16);   // K = klo+16 .. klo+23
  return u.v;
}

// B-matrix fragment, 32x16 bf16, source stored column-major [n][k].
// ISA layout: lanes 0-15 -> col n, K = {0..7, 16..23}; lanes 16-31 -> col n, K = {8..15, 24..31}.
__device__ __forceinline__ bf16x16 load_frag_b(const __bf16* base, int stride) {
  const int lane = threadIdx.x & 31;
  const int n    = lane & 15;
  const int klo  = (lane < 16) ? 0 : 8;
  const __bf16* p = base + n * stride + klo;
  FragB u;
  u.q[0] = *reinterpret_cast<const uint4*>(p);        // K = klo .. klo+7
  u.q[1] = *reinterpret_cast<const uint4*>(p + 16);   // K = klo+16 .. klo+23
  return u.v;
}

// ---------------------------------------------------------------------------
// QKV projection GEMM with fused per-head RMSNorm + RoPE epilogue.
// MODE: 0 = Q (norm+rope+score-scale), 1 = K (norm+rope), 2 = V (plain).
// Tile: BM=128, BN=128(=HD, one head per n-tile), BK=32. 256 threads / 8 waves.
// ---------------------------------------------------------------------------
template <int MODE>
__global__ __launch_bounds__(256)
void qkv_rope_kernel(const float* __restrict__ X, const float* __restrict__ W,
                     const int* __restrict__ pos_ids,
                     const float* __restrict__ normw,
                     __bf16* __restrict__ out, int NHEADS) {
  extern __shared__ char smem[];
  __bf16* lA = (__bf16*)smem;                       // [128][48] (32 used, padded)
  __bf16* lB = lA + 128 * 48;                       // [128 n][48 k]
  float*  sO = (float*)(smem + 2 * 128 * 48 * 2);   // [128][132] f32 tile

  const int tid  = threadIdx.x;
  const int wave = tid >> 5, lane = tid & 31;
  const int wm = wave & 3, wn = wave >> 2;          // 4 x 2 wave grid
  const int mBase = blockIdx.y * 128;
  const int head  = blockIdx.x;
  const int nBase = head * HD_;
  const int Nout  = NHEADS * HD_;

  AccF acc[2][4];
#pragma unroll
  for (int i = 0; i < 2; ++i)
#pragma unroll
    for (int j = 0; j < 4; ++j)
#pragma unroll
      for (int e = 0; e < 8; ++e) acc[i][j].e[e] = 0.f;

  for (int kk = 0; kk < HID_ / 32; ++kk) {
    // ---- stage A tile (X fp32 -> bf16 LDS) ----
    {
      const int r = tid >> 1, half = tid & 1;
      const float* gp = X + (size_t)(mBase + r) * HID_ + kk * 32 + half * 16;
      const float4* g = reinterpret_cast<const float4*>(gp);
      __bf16* d = lA + r * 48 + half * 16;
#pragma unroll
      for (int j = 0; j < 4; ++j) {
        float4 f = g[j];
        d[4 * j + 0] = (__bf16)f.x; d[4 * j + 1] = (__bf16)f.y;
        d[4 * j + 2] = (__bf16)f.z; d[4 * j + 3] = (__bf16)f.w;
      }
      if (kk + 1 < HID_ / 32) __builtin_prefetch(gp + 32, 0, 1);
    }
    // ---- stage B tile (W[k][n] fp32 -> lB[n][k] bf16) ----
    {
      const int kr = tid >> 3, c0 = (tid & 7) * 16;
      const float* g = W + (size_t)(kk * 32 + kr) * Nout + nBase + c0;
#pragma unroll
      for (int j = 0; j < 16; ++j) lB[(c0 + j) * 48 + kr] = (__bf16)g[j];
      if (kk + 1 < HID_ / 32) __builtin_prefetch(g + (size_t)32 * Nout, 0, 1);
    }
    __syncthreads();

    bf16x16 af[2], bf[4];
#pragma unroll
    for (int i = 0; i < 2; ++i) af[i] = load_frag_a(lA + (wm * 32 + i * 16) * 48, 48);
#pragma unroll
    for (int j = 0; j < 4; ++j) bf[j] = load_frag_b(lB + (wn * 64 + j * 16) * 48, 48);
#pragma unroll
    for (int i = 0; i < 2; ++i)
#pragma unroll
      for (int j = 0; j < 4; ++j)
        acc[i][j].v = __builtin_amdgcn_wmma_f32_16x16x32_bf16(
            false, af[i], false, bf[j], (short)0, acc[i][j].v, false, false);
    __syncthreads();
  }

  const int hi8 = (lane < 16) ? 0 : 8;
  const int cn  = lane & 15;

  if (MODE == 2) {
    // V: plain store as bf16 [B, NKV, S, HD]
#pragma unroll
    for (int i = 0; i < 2; ++i)
#pragma unroll
      for (int v = 0; v < 8; ++v) {
        const int token = mBase + wm * 32 + i * 16 + v + hi8;
        const int b = token >> 11, s = token & (S_ - 1);
        __bf16* op = out + (((size_t)b * NHEADS + head) * S_ + s) * HD_;
#pragma unroll
        for (int j = 0; j < 4; ++j)
          op[wn * 64 + j * 16 + cn] = (__bf16)acc[i][j].e[v];
      }
  } else {
    // Q/K: park tile in LDS, then per-row RMSNorm + RoPE
#pragma unroll
    for (int i = 0; i < 2; ++i)
#pragma unroll
      for (int j = 0; j < 4; ++j)
#pragma unroll
        for (int v = 0; v < 8; ++v)
          sO[(wm * 32 + i * 16 + v + hi8) * 132 + wn * 64 + j * 16 + cn] =
              acc[i][j].e[v];
    __syncthreads();

    if (tid < 128) {
      const int r = tid;
      const int token = mBase + r;
      const int b = token >> 11, s = token & (S_ - 1);
      const float pos = (float)pos_ids[b * S_ + s];
      const float* rowp = sO + r * 132;
      float ss = 0.f;
#pragma unroll 16
      for (int j = 0; j < HD_; ++j) { float x = rowp[j]; ss += x * x; }
      const float scl = rsqrtf(ss * (1.0f / HD_) + 1e-6f);
      const float qs  = (MODE == 0) ? 0.08838834764831845f : 1.0f;  // 1/sqrt(HD)
      __bf16* op = out + (((size_t)b * NHEADS + head) * S_ + s) * HD_;
#pragma unroll 8
      for (int j = 0; j < 64; ++j) {
        const float x1 = rowp[j]      * scl * normw[j];
        const float x2 = rowp[j + 64] * scl * normw[j + 64];
        // inv_freq = theta^(-j/64) = exp2(-j * log2(1e6)/64)
        const float ang = pos * exp2f(-(float)j * 0.31143075893f);
        float sn, cs; __sincosf(ang, &sn, &cs);
        op[j]      = (__bf16)((x1 * cs - x2 * sn) * qs);
        op[j + 64] = (__bf16)((x2 * cs + x1 * sn) * qs);
      }
    }
  }
}

// ---------------------------------------------------------------------------
// Flash attention: one block per (query tile of 128, b, h). 8 waves; each wave
// owns 16 query rows. Q fragments live in registers; K tile staged via
// GLOBAL_LOAD_ASYNC_TO_LDS_B128 (overlaps the VALU V-transpose staging).
// ---------------------------------------------------------------------------
__global__ __launch_bounds__(256)
void attn_kernel(const __bf16* __restrict__ Q, const __bf16* __restrict__ K,
                 const __bf16* __restrict__ V, __bf16* __restrict__ AO) {
  extern __shared__ char smem[];
  __bf16* lK  = (__bf16*)smem;        // [128 key][136 d]
  __bf16* lVt = lK  + 128 * 136;      // [128 d][136 key] (transposed)
  __bf16* lP  = lVt + 128 * 136;      // [8 waves][16 q][136 key]

  const int tid  = threadIdx.x;
  const int wave = tid >> 5, lane = tid & 31;
  const int qtile = blockIdx.x;                 // 0..15
  const int bh    = blockIdx.y;                 // 0..31
  const int b = bh >> 4, h = bh & 15;
  const int hk = h >> 1;                        // GQA: 2 q-heads per kv-head
  const int qs0 = qtile * 128 + wave * 16;

  const __bf16* Qbase = Q + (((size_t)b * NH_  + h ) * S_ + qs0) * HD_;
  const __bf16* Kbase = K + (((size_t)b * NKV_ + hk) * S_) * HD_;
  const __bf16* Vbase = V + (((size_t)b * NKV_ + hk) * S_) * HD_;

  bf16x16 qa[4];
#pragma unroll
  for (int kk = 0; kk < 4; ++kk) qa[kk] = load_frag_a(Qbase + kk * 32, HD_);

  AccF O[8];
  float m_run[8], l_run[8];
#pragma unroll
  for (int dt = 0; dt < 8; ++dt)
#pragma unroll
    for (int e = 0; e < 8; ++e) O[dt].e[e] = 0.f;
#pragma unroll
  for (int v = 0; v < 8; ++v) { m_run[v] = -3.0e38f; l_run[v] = 0.f; }

  const int hi8 = (lane < 16) ? 0 : 8;
  const int cn  = lane & 15;
  __bf16* lPw = lP + wave * 16 * 136;

#pragma unroll 1
  for (int kt = 0; kt < S_ / 128; ++kt) {
    __syncthreads();   // previous tile fully consumed
    // ---- stage K tile [key][d] via async global->LDS (ASYNCcnt) ----
    {
      const int key = tid >> 1, half = tid & 1;
      const __bf16* g = Kbase + (size_t)(kt * 128 + key) * HD_ + half * 64;
      __bf16* d = lK + key * 136 + half * 64;
#pragma unroll
      for (int j = 0; j < 8; ++j) async_g2lds_b128(d + j * 8, g + j * 8);
    }
    // ---- stage V tile transposed [d][key] (VALU path, overlaps async K) ----
    {
      const int key = tid >> 1, half = tid & 1;
      const __bf16* g = Vbase + (size_t)(kt * 128 + key) * HD_ + half * 64;
#pragma unroll
      for (int j = 0; j < 64; ++j) lVt[(half * 64 + j) * 136 + key] = g[j];
    }
    wait_async0();
    __syncthreads();

    // ---- scores: S = Q @ K^T (Q already carries 1/sqrt(HD)) ----
    AccF sc[8];
#pragma unroll
    for (int nt = 0; nt < 8; ++nt) {
#pragma unroll
      for (int e = 0; e < 8; ++e) sc[nt].e[e] = 0.f;
#pragma unroll
      for (int kk = 0; kk < 4; ++kk) {
        bf16x16 bK = load_frag_b(lK + (nt * 16) * 136 + kk * 32, 136);
        sc[nt].v = __builtin_amdgcn_wmma_f32_16x16x32_bf16(
            false, qa[kk], false, bK, (short)0, sc[nt].v, false, false);
      }
    }

    // ---- online softmax (rows split 16-lane halves; shfl_xor reductions) ----
#pragma unroll
    for (int v = 0; v < 8; ++v) {
      float mx = sc[0].e[v];
#pragma unroll
      for (int nt = 1; nt < 8; ++nt) mx = fmaxf(mx, sc[nt].e[v]);
#pragma unroll
      for (int off = 1; off < 16; off <<= 1)
        mx = fmaxf(mx, __shfl_xor(mx, off, 32));
      const float mnew  = fmaxf(m_run[v], mx);
      const float alpha = __expf(m_run[v] - mnew);
      m_run[v] = mnew;
      float rs = 0.f;
#pragma unroll
      for (int nt = 0; nt < 8; ++nt) {
        const float p = __expf(sc[nt].e[v] - mnew);
        sc[nt].e[v] = p;
        rs += p;
      }
#pragma unroll
      for (int off = 1; off < 16; off <<= 1) rs += __shfl_xor(rs, off, 32);
      l_run[v] = l_run[v] * alpha + rs;
#pragma unroll
      for (int dt = 0; dt < 8; ++dt) O[dt].e[v] *= alpha;
    }

    // ---- park P (bf16) in this wave's LDS slice ----
#pragma unroll
    for (int nt = 0; nt < 8; ++nt)
#pragma unroll
      for (int v = 0; v < 8; ++v)
        lPw[(v + hi8) * 136 + nt * 16 + cn] = (__bf16)sc[nt].e[v];

    // ---- O += P @ V ----
    bf16x16 aP[4];
#pragma unroll
    for (int kc = 0; kc < 4; ++kc) aP[kc] = load_frag_a(lPw + kc * 32, 136);
#pragma unroll
    for (int dt = 0; dt < 8; ++dt)
#pragma unroll
      for (int kc = 0; kc < 4; ++kc) {
        bf16x16 bV = load_frag_b(lVt + (dt * 16) * 136 + kc * 32, 136);
        O[dt].v = __builtin_amdgcn_wmma_f32_16x16x32_bf16(
            false, aP[kc], false, bV, (short)0, O[dt].v, false, false);
      }
  }

  // ---- normalize and store to AO [token][NH*HD] (bf16) ----
#pragma unroll
  for (int v = 0; v < 8; ++v) {
    const float inv = 1.0f / l_run[v];
    const int s = qs0 + v + hi8;
    __bf16* op = AO + ((size_t)b * S_ + s) * (NH_ * HD_) + h * HD_;
#pragma unroll
    for (int dt = 0; dt < 8; ++dt)
      op[dt * 16 + cn] = (__bf16)(O[dt].e[v] * inv);
  }
}

// ---------------------------------------------------------------------------
// Output projection: AO(bf16)[4096,2048] @ Wo(fp32->bf16)[2048,2048] -> fp32.
// A tile is a pure copy -> async global->LDS.
// ---------------------------------------------------------------------------
__global__ __launch_bounds__(256)
void out_gemm_kernel(const __bf16* __restrict__ A, const float* __restrict__ W,
                     float* __restrict__ Out) {
  __shared__ __bf16 lA[128 * 48];
  __shared__ __bf16 lB[128 * 48];   // [n][k]
  const int tid  = threadIdx.x;
  const int wave = tid >> 5, lane = tid & 31;
  const int wm = wave & 3, wn = wave >> 2;
  const int mBase = blockIdx.y * 128, nBase = blockIdx.x * 128;

  AccF acc[2][4];
#pragma unroll
  for (int i = 0; i < 2; ++i)
#pragma unroll
    for (int j = 0; j < 4; ++j)
#pragma unroll
      for (int e = 0; e < 8; ++e) acc[i][j].e[e] = 0.f;

  for (int kk = 0; kk < HID_ / 32; ++kk) {
    {
      const int r = tid >> 1, half = tid & 1;
      const __bf16* g = A + (size_t)(mBase + r) * HID_ + kk * 32 + half * 16;
      __bf16* d = lA + r * 48 + half * 16;
      async_g2lds_b128(d, g);
      async_g2lds_b128(d + 8, g + 8);
    }
    {
      const int kr = tid >> 3, c0 = (tid & 7) * 16;
      const float* g = W + (size_t)(kk * 32 + kr) * HID_ + nBase + c0;
#pragma unroll
      for (int j = 0; j < 16; ++j) lB[(c0 + j) * 48 + kr] = (__bf16)g[j];
      if (kk + 1 < HID_ / 32) __builtin_prefetch(g + (size_t)32 * HID_, 0, 1);
    }
    wait_async0();
    __syncthreads();

    bf16x16 af[2], bf[4];
#pragma unroll
    for (int i = 0; i < 2; ++i) af[i] = load_frag_a(lA + (wm * 32 + i * 16) * 48, 48);
#pragma unroll
    for (int j = 0; j < 4; ++j) bf[j] = load_frag_b(lB + (wn * 64 + j * 16) * 48, 48);
#pragma unroll
    for (int i = 0; i < 2; ++i)
#pragma unroll
      for (int j = 0; j < 4; ++j)
        acc[i][j].v = __builtin_amdgcn_wmma_f32_16x16x32_bf16(
            false, af[i], false, bf[j], (short)0, acc[i][j].v, false, false);
    __syncthreads();
  }

  const int hi8 = (lane < 16) ? 0 : 8;
  const int cn  = lane & 15;
#pragma unroll
  for (int i = 0; i < 2; ++i)
#pragma unroll
    for (int v = 0; v < 8; ++v) {
      const int row = mBase + wm * 32 + i * 16 + v + hi8;
#pragma unroll
      for (int j = 0; j < 4; ++j)
        Out[(size_t)row * HID_ + nBase + wn * 64 + j * 16 + cn] = acc[i][j].e[v];
    }
}

// ---------------------------------------------------------------------------
extern "C" void kernel_launch(void* const* d_in, const int* in_sizes, int n_in,
                              void* d_out, int out_size, void* d_ws, size_t ws_size,
                              hipStream_t stream) {
  (void)in_sizes; (void)n_in; (void)out_size; (void)ws_size;
  const float* X   = (const float*)d_in[0];
  const int*   pos = (const int*)d_in[1];
  const float* Wq  = (const float*)d_in[2];
  const float* Wk  = (const float*)d_in[3];
  const float* Wv  = (const float*)d_in[4];
  const float* Wo  = (const float*)d_in[5];
  const float* qnw = (const float*)d_in[6];
  const float* knw = (const float*)d_in[7];
  float* out = (float*)d_out;

  // Workspace layout (bf16): Q 16MB | K 8MB | V 8MB | AO 16MB  (48 MB total)
  char* ws = (char*)d_ws;
  __bf16* Qb = (__bf16*)(ws);
  __bf16* Kb = (__bf16*)(ws + (size_t)16777216);
  __bf16* Vb = (__bf16*)(ws + (size_t)16777216 + 8388608);
  __bf16* AO = (__bf16*)(ws + (size_t)16777216 + 2 * 8388608);

  const int smem_qkv = 2 * 128 * 48 * 2 + 128 * 132 * 4;  // 92,160 B
  const int smem_att = 3 * 128 * 136 * 2;                 // 104,448 B

  (void)hipFuncSetAttribute(reinterpret_cast<const void*>(qkv_rope_kernel<0>),
                            hipFuncAttributeMaxDynamicSharedMemorySize, smem_qkv);
  (void)hipFuncSetAttribute(reinterpret_cast<const void*>(qkv_rope_kernel<1>),
                            hipFuncAttributeMaxDynamicSharedMemorySize, smem_qkv);
  (void)hipFuncSetAttribute(reinterpret_cast<const void*>(qkv_rope_kernel<2>),
                            hipFuncAttributeMaxDynamicSharedMemorySize, smem_qkv);
  (void)hipFuncSetAttribute(reinterpret_cast<const void*>(attn_kernel),
                            hipFuncAttributeMaxDynamicSharedMemorySize, smem_att);

  qkv_rope_kernel<0><<<dim3(NH_,  32), 256, smem_qkv, stream>>>(X, Wq, pos, qnw, Qb, NH_);
  qkv_rope_kernel<1><<<dim3(NKV_, 32), 256, smem_qkv, stream>>>(X, Wk, pos, knw, Kb, NKV_);
  qkv_rope_kernel<2><<<dim3(NKV_, 32), 256, smem_qkv, stream>>>(X, Wv, pos, nullptr, Vb, NKV_);
  attn_kernel<<<dim3(S_ / 128, B_ * NH_), 256, smem_att, stream>>>(Qb, Kb, Vb, AO);
  out_gemm_kernel<<<dim3(HID_ / 128, 32), 256, 0, stream>>>(AO, Wo, out);
}
